// S5Block_40578851012648
// MI455X (gfx1250) — matrix operand order repaired and checked by reading, not compiled, not run on hardware
//
#include <hip/hip_runtime.h>
#include <hip/hip_bf16.h>

// ---------------------------------------------------------------------------
// Types for CDNA5 WMMA (wave32): v_wmma_f32_16x16x32_bf16
// ---------------------------------------------------------------------------
typedef __bf16 bf16_t;
typedef __attribute__((ext_vector_type(16))) __bf16 v16bf;
typedef __attribute__((ext_vector_type(8)))  __bf16 v8bf;
typedef __attribute__((ext_vector_type(8)))  float  v8f;

__device__ __forceinline__ bf16_t f2bf(float f) {
    unsigned u = __float_as_uint(f);
    u += 0x7FFFu + ((u >> 16) & 1u);            // round-to-nearest-even
    union { unsigned short s; __bf16 b; } cv;
    cv.s = (unsigned short)(u >> 16);
    return cv.b;
}

__device__ __forceinline__ float gelu_exact(float x) {
    return 0.5f * x * (1.0f + erff(x * 0.70710678118654752f));
}

// ---------------------------------------------------------------------------
// Small elementwise / prep kernels
// ---------------------------------------------------------------------------
__global__ void k_lambda(const float* __restrict__ Lre, const float* __restrict__ Lim,
                         const float* __restrict__ lstep,
                         float* __restrict__ lbr, float* __restrict__ lbi,
                         float* __restrict__ cfr, float* __restrict__ cfi) {
    int p = threadIdx.x;                          // 512 states
    float dt = expf(lstep[p]);
    float lr = Lre[p], li = Lim[p];
    float er = expf(lr * dt);
    float th = li * dt;
    float Lbr = er * cosf(th);
    float Lbi = er * sinf(th);
    float den = lr * lr + li * li;
    float nr = Lbr - 1.0f, ni = Lbi;
    cfr[p] = (nr * lr + ni * li) / den;           // coef = (Lam_bar-1)/Lam
    cfi[p] = (ni * lr - nr * li) / den;
    lbr[p] = Lbr; lbi[p] = Lbi;
}

// Bmat[N=2P][K=H]: rows 0..511 = Re(coef*Bc)[p][h], rows 512..1023 = Im
__global__ void k_make_bmat(const float* __restrict__ Bre, const float* __restrict__ Bim,
                            const float* __restrict__ cfr, const float* __restrict__ cfi,
                            bf16_t* __restrict__ Bmat) {
    int i = blockIdx.x * 256 + threadIdx.x;       // P*H = 262144
    if (i < 512 * 512) {
        int p = i >> 9, h = i & 511;
        float br = Bre[i], bi = Bim[i];
        float cr = cfr[p], ci = cfi[p];
        Bmat[(size_t)p * 512 + h]          = f2bf(cr * br - ci * bi);
        Bmat[(size_t)(512 + p) * 512 + h]  = f2bf(cr * bi + ci * br);
    }
}

// Cmat[N=H][K=2P]: row h: cols 0..511 = 2*C_re[h][p], cols 512..1023 = -2*C_im[h][p]
__global__ void k_make_cmat(const float* __restrict__ Cre, const float* __restrict__ Cim,
                            bf16_t* __restrict__ Cmat) {
    int i = blockIdx.x * 256 + threadIdx.x;       // H*P
    if (i < 512 * 512) {
        int h = i >> 9, p = i & 511;
        Cmat[(size_t)h * 1024 + p]        = f2bf( 2.0f * Cre[i]);
        Cmat[(size_t)h * 1024 + 512 + p]  = f2bf(-2.0f * Cim[i]);
    }
}

__global__ void k_conv_bf16(const float* __restrict__ in, bf16_t* __restrict__ out, int n) {
    int i = blockIdx.x * 256 + threadIdx.x;
    if (i < n) out[i] = f2bf(in[i]);
}

// LayerNorm over H=512; one row per 256-thread block; emits f32 + bf16
__global__ void k_ln(const float* __restrict__ in, const float* __restrict__ w,
                     const float* __restrict__ b,
                     float* __restrict__ outf, bf16_t* __restrict__ outb) {
    __shared__ float s1[8], s2[8];
    size_t row = blockIdx.x;
    const float* r = in + row * 512;
    int t = threadIdx.x;
    float x0 = r[t], x1 = r[t + 256];
    float s = x0 + x1, q = x0 * x0 + x1 * x1;
    for (int o = 16; o > 0; o >>= 1) { s += __shfl_xor(s, o, 32); q += __shfl_xor(q, o, 32); }
    int wv = t >> 5, ln = t & 31;
    if (ln == 0) { s1[wv] = s; s2[wv] = q; }
    __syncthreads();
    if (wv == 0) {
        float a = (ln < 8) ? s1[ln] : 0.0f, c = (ln < 8) ? s2[ln] : 0.0f;
        for (int o = 4; o > 0; o >>= 1) { a += __shfl_xor(a, o, 32); c += __shfl_xor(c, o, 32); }
        if (ln == 0) { s1[0] = a; s2[0] = c; }
    }
    __syncthreads();
    float mu  = s1[0] * (1.0f / 512.0f);
    float var = s2[0] * (1.0f / 512.0f) - mu * mu;
    float inv = rsqrtf(var + 1e-5f);
    float y0 = (x0 - mu) * inv * w[t]       + b[t];
    float y1 = (x1 - mu) * inv * w[t + 256] + b[t + 256];
    outf[row * 512 + t]       = y0;  outb[row * 512 + t]       = f2bf(y0);
    outf[row * 512 + t + 256] = y1;  outb[row * 512 + t + 256] = f2bf(y1);
}

// ---------------------------------------------------------------------------
// Chunked parallel scan: x_t = Lam_bar * x_{t-1} + Bu_t   (per batch b, state p)
// bu layout: [16384 rows][1024] (cols 0..511 re, 512..1023 im); scanned in place
// ---------------------------------------------------------------------------
__global__ void k_scan_local(float* __restrict__ bu,
                             const float* __restrict__ lbr, const float* __restrict__ lbi,
                             float* __restrict__ lastr, float* __restrict__ lasti) {
    int c = blockIdx.x, bb = blockIdx.y, p = threadIdx.x;   // 8 chunks x 4 batches x 512 p
    float ar = lbr[p], ai = lbi[p];
    float xr = 0.f, xi = 0.f;
    size_t base = ((size_t)bb * 4096 + (size_t)c * 512) * 1024;
    for (int j = 0; j < 512; ++j) {
        size_t o = base + (size_t)j * 1024;
        float ur = bu[o + p], ui = bu[o + 512 + p];
        float nr = ar * xr - ai * xi + ur;
        float ni = ar * xi + ai * xr + ui;
        xr = nr; xi = ni;
        bu[o + p] = xr; bu[o + 512 + p] = xi;
    }
    int cid = bb * 8 + c;
    lastr[cid * 512 + p] = xr; lasti[cid * 512 + p] = xi;
}

__global__ void k_scan_carry(const float* __restrict__ lbr, const float* __restrict__ lbi,
                             const float* __restrict__ lastr, const float* __restrict__ lasti,
                             float* __restrict__ cinr, float* __restrict__ cini) {
    int bb = blockIdx.x, p = threadIdx.x;
    float pr = lbr[p], pi = lbi[p];                 // Lam_bar^512 via 9 squarings
    for (int i = 0; i < 9; ++i) { float nr = pr*pr - pi*pi, ni = 2.f*pr*pi; pr = nr; pi = ni; }
    float cr = 0.f, ci = 0.f;
    for (int c = 0; c < 8; ++c) {
        int cid = bb * 8 + c;
        cinr[cid * 512 + p] = cr; cini[cid * 512 + p] = ci;
        float nr = pr * cr - pi * ci + lastr[cid * 512 + p];
        float ni = pr * ci + pi * cr + lasti[cid * 512 + p];
        cr = nr; ci = ni;
    }
}

__global__ void k_scan_fix(const float* __restrict__ bu,
                           const float* __restrict__ lbr, const float* __restrict__ lbi,
                           const float* __restrict__ cinr, const float* __restrict__ cini,
                           bf16_t* __restrict__ xsb) {
    int c = blockIdx.x, bb = blockIdx.y, p = threadIdx.x;
    int cid = bb * 8 + c;
    float ar = lbr[p], ai = lbi[p];
    float cr = cinr[cid * 512 + p], ci = cini[cid * 512 + p];
    size_t base = ((size_t)bb * 4096 + (size_t)c * 512) * 1024;
    for (int j = 0; j < 512; ++j) {
        float nr = ar * cr - ai * ci, ni = ar * ci + ai * cr;   // A^{j+1} * carry
        cr = nr; ci = ni;
        size_t o = base + (size_t)j * 1024;
        xsb[o + p]       = f2bf(bu[o + p] + cr);
        xsb[o + 512 + p] = f2bf(bu[o + 512 + p] + ci);
    }
}

// ---------------------------------------------------------------------------
// WMMA GEMM: C[M][N] = A[M][K] * B[N][K]^T (+resid). bf16 in, f32 accumulate.
// Block tile 128x128, 8 waves of wave32, each wave 32x64 (2x4 WMMA tiles).
// Staging: GLOBAL_LOAD_ASYNC_TO_LDS_B128 (ASYNCcnt) with double-buffered LDS.
// Per-wave ordering: s_wait_asynccnt 0 -> barrier publishes a complete stage.
// ---------------------------------------------------------------------------
__device__ __forceinline__ v16bf load_frag(const bf16_t* base) {
    v8bf lo = *(const v8bf*)(base);
    v8bf hi = *(const v8bf*)(base + 16);
    return __builtin_shufflevector(lo, hi, 0,1,2,3,4,5,6,7,8,9,10,11,12,13,14,15);
}
__device__ __forceinline__ v16bf load_fragB(const bf16_t* base) {
    v8bf lo = *(const v8bf*)(base);
    v8bf hi = *(const v8bf*)(base + 16);
    return __builtin_shufflevector(lo, hi, 0,1,2,3,4,5,6,7,8,9,10,11,12,13,14,15);
}

__device__ __forceinline__ void async_b128(unsigned lds_off, uint64_t gaddr) {
    asm volatile("global_load_async_to_lds_b128 %0, %1, off"
                 :: "v"(lds_off), "v"(gaddr) : "memory");
}
__device__ __forceinline__ void wait_async0() {
    asm volatile("s_wait_asynccnt 0x0" ::: "memory");
}

__global__ void k_gemm(const bf16_t* __restrict__ A, const bf16_t* __restrict__ B,
                       float* __restrict__ C, const float* __restrict__ resid,
                       int lda, int ldb, int ldc, int K) {
    __shared__ __align__(16) bf16_t As[2][128 * 40];   // 32 K + 8 pad, 80B pitch
    __shared__ __align__(16) bf16_t Bs[2][128 * 40];
    const int m0 = blockIdx.y * 128;
    const int n0 = blockIdx.x * 128;
    const int tid  = threadIdx.x;
    const int lane = tid & 31;
    const int wave = tid >> 5;
    const int wm = wave & 3, wn = wave >> 2;
    const int rl = lane & 15;
    const int kbaseA = (lane >> 4) * 8;             // A frag: K {0-7,16-23} / {8-15,24-31}
    const int kbaseB = (lane >> 4) * 8;             // B frag: K {0-7,16-23} / {8-15,24-31}

    // staging indices: 128 rows x 4 uint4 per row = 512 uint4 each for A and B
    const int r0 = tid >> 2,  c0 = tid & 3;         // rows 0..63
    const int r1 = r0 + 64,   c1 = c0;              // rows 64..127

    v8f acc[2][4];
    v8f z = {};
    for (int mt = 0; mt < 2; ++mt)
        for (int nt = 0; nt < 4; ++nt) acc[mt][nt] = z;

    auto stage = [&](int k0, int buf) {
        async_b128((unsigned)(uintptr_t)&As[buf][r0 * 40 + c0 * 8],
                   (uint64_t)(uintptr_t)&A[(size_t)(m0 + r0) * lda + k0 + c0 * 8]);
        async_b128((unsigned)(uintptr_t)&As[buf][r1 * 40 + c1 * 8],
                   (uint64_t)(uintptr_t)&A[(size_t)(m0 + r1) * lda + k0 + c1 * 8]);
        async_b128((unsigned)(uintptr_t)&Bs[buf][r0 * 40 + c0 * 8],
                   (uint64_t)(uintptr_t)&B[(size_t)(n0 + r0) * ldb + k0 + c0 * 8]);
        async_b128((unsigned)(uintptr_t)&Bs[buf][r1 * 40 + c1 * 8],
                   (uint64_t)(uintptr_t)&B[(size_t)(n0 + r1) * ldb + k0 + c1 * 8]);
    };

    const int nsteps = K >> 5;
    stage(0, 0);
    for (int s = 0; s < nsteps; ++s) {
        const int cur = s & 1;
        wait_async0();           // my async writes into buf[cur] are in LDS
        __syncthreads();         // => everyone's are; buf[cur^1] no longer read
        if (s + 1 < nsteps) stage((s + 1) << 5, cur ^ 1);

        const bf16_t* as = As[cur];
        const bf16_t* bs = Bs[cur];
        v16bf a0 = load_frag (&as[(wm * 32 +  0 + rl) * 40 + kbaseA]);
        v16bf a1 = load_frag (&as[(wm * 32 + 16 + rl) * 40 + kbaseA]);
        v16bf b0 = load_fragB(&bs[(wn * 64 +  0 + rl) * 40 + kbaseB]);
        v16bf b1 = load_fragB(&bs[(wn * 64 + 16 + rl) * 40 + kbaseB]);
        v16bf b2 = load_fragB(&bs[(wn * 64 + 32 + rl) * 40 + kbaseB]);
        v16bf b3 = load_fragB(&bs[(wn * 64 + 48 + rl) * 40 + kbaseB]);
        acc[0][0] = __builtin_amdgcn_wmma_f32_16x16x32_bf16(false, a0, false, b0, (short)0, acc[0][0], false, false);
        acc[0][1] = __builtin_amdgcn_wmma_f32_16x16x32_bf16(false, a0, false, b1, (short)0, acc[0][1], false, false);
        acc[0][2] = __builtin_amdgcn_wmma_f32_16x16x32_bf16(false, a0, false, b2, (short)0, acc[0][2], false, false);
        acc[0][3] = __builtin_amdgcn_wmma_f32_16x16x32_bf16(false, a0, false, b3, (short)0, acc[0][3], false, false);
        acc[1][0] = __builtin_amdgcn_wmma_f32_16x16x32_bf16(false, a1, false, b0, (short)0, acc[1][0], false, false);
        acc[1][1] = __builtin_amdgcn_wmma_f32_16x16x32_bf16(false, a1, false, b1, (short)0, acc[1][1], false, false);
        acc[1][2] = __builtin_amdgcn_wmma_f32_16x16x32_bf16(false, a1, false, b2, (short)0, acc[1][2], false, false);
        acc[1][3] = __builtin_amdgcn_wmma_f32_16x16x32_bf16(false, a1, false, b3, (short)0, acc[1][3], false, false);
    }

    // C/D layout: VGPR i -> M=i (lanes 0-15) / M=i+8 (lanes 16-31); N = lane&15
    const int rbase = m0 + wm * 32 + ((lane >= 16) ? 8 : 0);
    const int cbase = n0 + wn * 64 + (lane & 15);
    for (int mt = 0; mt < 2; ++mt)
        for (int nt = 0; nt < 4; ++nt)
            for (int i = 0; i < 8; ++i) {
                int rr = rbase + mt * 16 + i;
                int cc = cbase + nt * 16;
                float v = acc[mt][nt][i];
                if (resid) v += resid[(size_t)rr * ldc + cc];
                C[(size_t)rr * ldc + cc] = v;
            }
}

// ---------------------------------------------------------------------------
// Epilogue elementwise kernels
// ---------------------------------------------------------------------------
__global__ void k_post_s5(const float* __restrict__ y, float* __restrict__ fx,
                          const float* __restrict__ D, int n) {
    int i = blockIdx.x * 256 + threadIdx.x;         // n = 16384*512
    if (i < n) {
        int h = i & 511;
        float u = fx[i];
        float v = y[i] + D[h] * u;
        fx[i] = gelu_exact(v) + u;                  // h = gelu(y) + res, in place
    }
}

__global__ void k_geglu(const float* __restrict__ e, bf16_t* __restrict__ ff, int n) {
    int i = blockIdx.x * 256 + threadIdx.x;         // n = 16384*512
    if (i < n) {
        int r = i >> 9, c = i & 511;
        float a = e[(size_t)r * 1024 + c];
        float g = e[(size_t)r * 1024 + 512 + c];
        ff[i] = f2bf(a * gelu_exact(g));
    }
}

// ---------------------------------------------------------------------------
// Launch: B=4, L=4096, H=512, P=512, rows = 16384
// ---------------------------------------------------------------------------
extern "C" void kernel_launch(void* const* d_in, const int* in_sizes, int n_in,
                              void* d_out, int out_size, void* d_ws, size_t ws_size,
                              hipStream_t stream) {
    const float* x      = (const float*)d_in[0];
    const float* ln1w   = (const float*)d_in[1];
    const float* ln1b   = (const float*)d_in[2];
    const float* Lre    = (const float*)d_in[3];
    const float* Lim    = (const float*)d_in[4];
    const float* Bre    = (const float*)d_in[5];
    const float* Bim    = (const float*)d_in[6];
    const float* Cre    = (const float*)d_in[7];
    const float* Cim    = (const float*)d_in[8];
    const float* Dp     = (const float*)d_in[9];
    const float* lstep  = (const float*)d_in[10];
    const float* ln2w   = (const float*)d_in[11];
    const float* ln2b   = (const float*)d_in[12];
    const float* Wenc   = (const float*)d_in[13];
    const float* Wdec   = (const float*)d_in[14];
    float* out = (float*)d_out;

    char* ws = (char*)d_ws;
    float*  fx   = (float*)(ws);                          // 32 MB (later h, in place)
    bf16_t* fxb  = (bf16_t*)(ws + (32ull << 20));         // 16 MB
    float*  big  = (float*)(ws + (48ull << 20));          // 64 MB  Bu/xs f32, later e f32
    bf16_t* xsb  = (bf16_t*)(ws + (112ull << 20));        // 32 MB  xs bf16; later fx2 bf16
    bf16_t* ffb  = (bf16_t*)(ws + (128ull << 20));        // 16 MB  ff bf16
    float*  yraw = (float*)(ws + (144ull << 20));         // 32 MB  y raw; later fx2 f32
    char*   wb   = ws + (176ull << 20);
    bf16_t* Bmat = (bf16_t*)(wb);                         // 1 MB  [1024][512]
    bf16_t* Cmat = (bf16_t*)(wb + (1ull << 20));          // 1 MB  [512][1024]
    bf16_t* WencB= (bf16_t*)(wb + (2ull << 20));          // 1 MB  [1024][512]
    bf16_t* WdecB= (bf16_t*)(wb + (3ull << 20));          // 0.5MB [512][512]
    float* lbr   = (float*)(wb + (4ull << 20));
    float* lbi   = lbr + 512;
    float* cfr   = lbi + 512;
    float* cfi   = cfr + 512;
    float* lastr = cfi + 512;                             // 32*512 each
    float* lasti = lastr + 32 * 512;
    float* cinr  = lasti + 32 * 512;
    float* cini  = cinr + 32 * 512;

    // --- parameter prep (bf16 weights in [N][K] layout) ---
    k_lambda   <<<1, 512, 0, stream>>>(Lre, Lim, lstep, lbr, lbi, cfr, cfi);
    k_make_bmat<<<1024, 256, 0, stream>>>(Bre, Bim, cfr, cfi, Bmat);
    k_make_cmat<<<1024, 256, 0, stream>>>(Cre, Cim, Cmat);
    k_conv_bf16<<<2048, 256, 0, stream>>>(Wenc, WencB, 1024 * 512);
    k_conv_bf16<<<1024, 256, 0, stream>>>(Wdec, WdecB, 512 * 512);

    // --- ln1 ---
    k_ln<<<16384, 256, 0, stream>>>(x, ln1w, ln1b, fx, fxb);

    // --- Bu = fx @ Bmat^T   [16384 x 1024] ---
    k_gemm<<<dim3(8, 128), 256, 0, stream>>>(fxb, Bmat, big, nullptr, 512, 512, 1024, 512);

    // --- associative scan along L (chunked) ---
    k_scan_local<<<dim3(8, 4), 512, 0, stream>>>(big, lbr, lbi, lastr, lasti);
    k_scan_carry<<<4, 512, 0, stream>>>(lbr, lbi, lastr, lasti, cinr, cini);
    k_scan_fix  <<<dim3(8, 4), 512, 0, stream>>>(big, lbr, lbi, cinr, cini, xsb);

    // --- y = xs @ Cmat^T   [16384 x 512] ---
    k_gemm<<<dim3(4, 128), 256, 0, stream>>>(xsb, Cmat, yraw, nullptr, 1024, 1024, 512, 1024);

    // --- h = gelu(y + D*u) + res (in place over fx) ---
    k_post_s5<<<32768, 256, 0, stream>>>(yraw, fx, Dp, 16384 * 512);

    // --- ln2: fx2 (f32 into yraw region, bf16 into xsb region) ---
    k_ln<<<16384, 256, 0, stream>>>(fx, ln2w, ln2b, yraw, xsb);

    // --- e = fx2 @ Wenc^T   [16384 x 1024] ---
    k_gemm<<<dim3(8, 128), 256, 0, stream>>>(xsb, WencB, big, nullptr, 512, 512, 1024, 512);

    // --- GEGLU: ff = a * gelu(g) ---
    k_geglu<<<32768, 256, 0, stream>>>(big, ffb, 16384 * 512);

    // --- out = ff @ Wdec^T + fx2 ---
    k_gemm<<<dim3(4, 128), 256, 0, stream>>>(ffb, WdecB, out, yraw, 512, 512, 512, 512);
}
